// TMSAG_70325794505342
// MI455X (gfx1250) — matrix operand order repaired and checked by reading, not compiled, not run on hardware
//
#include <hip/hip_runtime.h>
#include <hip/hip_bf16.h>

typedef _Float16 half_t;
typedef __attribute__((ext_vector_type(8)))  _Float16 v8h;
typedef __attribute__((ext_vector_type(16))) _Float16 v16h;
typedef __attribute__((ext_vector_type(8)))  float    v8f;

#define NWIN 864
#define NTOK 110592   // 12*96*96
#define GEMM_BIAS 0
#define GEMM_GATE 1
#define GEMM_RESBIAS 2

// ---------------- WMMA fragment helpers (CDNA5 16x16x32 f16 layouts) -------

__device__ inline v8f vzero8() { v8f z; for (int i = 0; i < 8; ++i) z[i] = 0.f; return z; }

// A matrix 16x32 f16: lane L<16 holds row L, K in {kb..kb+7, 16+kb..16+kb+7}, kb = (L<16)?0:8
__device__ inline v16h frag_A(const half_t* p, int ld, int lane) {
  int row = lane & 15;
  int kb  = (lane < 16) ? 0 : 8;
  const half_t* q = p + row * ld + kb;
  v8h lo = *(const v8h*)q;
  v8h hi = *(const v8h*)(q + 16);
  v16h r;
  for (int i = 0; i < 8; ++i) { r[i] = lo[i]; r[i + 8] = hi[i]; }
  return r;
}
// B matrix 32x16 f16 given W stored row-major [n][k] (B[k][n] = W[n][k]):
// lane holds column n = L%16, K in {kb..kb+7, 16+kb..16+kb+7}, kb = (L<16)?0:8
__device__ inline v16h frag_B(const half_t* p, int ld, int lane) {
  int n  = lane & 15;
  int kb = (lane < 16) ? 0 : 8;
  const half_t* q = p + n * ld + kb;
  v8h lo = *(const v8h*)q;
  v8h hi = *(const v8h*)(q + 16);
  v16h r;
  for (int i = 0; i < 8; ++i) { r[i] = lo[i]; r[i + 8] = hi[i]; }
  return r;
}
// C/D 16x16 f32: lane: col = L%16, vgpr r -> row r + 8*(L>=16)
__device__ inline v8f frag_C(const float* p, int ld, int lane) {
  int n = lane & 15, rb = (lane < 16) ? 0 : 8;
  v8f c;
  for (int r = 0; r < 8; ++r) c[r] = p[(rb + r) * ld + n];
  return c;
}
__device__ inline void store_D(float* p, int ld, int lane, v8f d) {
  int n = lane & 15, rb = (lane < 16) ? 0 : 8;
  for (int r = 0; r < 8; ++r) p[(rb + r) * ld + n] = d[r];
}
__device__ inline v8f wmma16(v16h a, v16h b, v8f c) {
  return __builtin_amdgcn_wmma_f32_16x16x32_f16(false, a, false, b, (short)0, c, false, false);
}

__device__ inline int group3(int g, int b0, int b1) { return g < b0 ? 0 : (g < b1 ? 1 : 2); }

// ---------------- layout / weight packing ----------------------------------

__global__ void k_transpose_in(const float* __restrict__ xin, float* __restrict__ xtok) {
  size_t i = (size_t)blockIdx.x * 256 + threadIdx.x;
  if (i >= (size_t)120 * NTOK) return;
  size_t c = i / NTOK, t = i % NTOK;
  xtok[t * 120 + c] = xin[i];
}
__global__ void k_transpose_out(const float* __restrict__ xtok, float* __restrict__ out) {
  size_t i = (size_t)blockIdx.x * 256 + threadIdx.x;
  if (i >= (size_t)120 * NTOK) return;
  size_t c = i / NTOK, t = i % NTOK;
  out[i] = xtok[t * 120 + c];
}

// qkv weight (360,120) -> per-head padded (576,128): row = h*96 + s*32 + j
__global__ void k_pack_qkv(const float* __restrict__ src, half_t* __restrict__ dst) {
  int i = blockIdx.x * 256 + threadIdx.x;
  if (i >= 576 * 128) return;
  int r = i >> 7, k = i & 127;
  int h = r / 96, s = (r % 96) / 32, j = r % 32;
  float v = (j < 20 && k < 120) ? src[(s * 120 + h * 20 + j) * 120 + k] : 0.f;
  dst[i] = (half_t)v;
}
// proj weight (120,240) -> (128,256): cols 0..119 <- in 0..119, 128..247 <- in 120..239
__global__ void k_pack_proj(const float* __restrict__ src, half_t* __restrict__ dst) {
  int i = blockIdx.x * 256 + threadIdx.x;
  if (i >= 128 * 256) return;
  int nn = i >> 8, k = i & 255;
  float v = 0.f;
  if (nn < 120) {
    if (k < 120) v = src[nn * 240 + k];
    else if (k >= 128 && k < 248) v = src[nn * 240 + (k - 8)];
  }
  dst[i] = (half_t)v;
}
// fc11/fc12 (240,120) -> interleaved 16-row blocks (512,128): even blk = fc11, odd = fc12
__global__ void k_pack_w1c(const float* __restrict__ f11, const float* __restrict__ f12,
                           half_t* __restrict__ dst) {
  int i = blockIdx.x * 256 + threadIdx.x;
  if (i >= 512 * 128) return;
  int r = i >> 7, k = i & 127;
  int blk = r >> 4, j = r & 15;
  int hid = (blk >> 1) * 16 + j;
  const float* s = (blk & 1) ? f12 : f11;
  float v = (hid < 240 && k < 120) ? s[hid * 120 + k] : 0.f;
  dst[i] = (half_t)v;
}
// fc2 weight (120,240) -> (128,256)
__global__ void k_pack_fc2(const float* __restrict__ src, half_t* __restrict__ dst) {
  int i = blockIdx.x * 256 + threadIdx.x;
  if (i >= 128 * 256) return;
  int nn = i >> 8, k = i & 255;
  float v = (nn < 120 && k < 240) ? src[nn * 240 + k] : 0.f;
  dst[i] = (half_t)v;
}

// ---------------- LN + shift + window partition (+sine pos) ----------------

__global__ void k_ln_part(const float* __restrict__ x, const float* __restrict__ w,
                          const float* __restrict__ b, int sd, int sh, int sw,
                          half_t* __restrict__ xw, half_t* __restrict__ xwp) {
  int wi = blockIdx.x, tk = threadIdx.x;                 // 864 blocks x 128 threads
  int wd = wi / 144, wh = (wi / 12) % 12, ww = wi % 12;
  int id = tk >> 6, ih = (tk >> 3) & 7, iw = tk & 7;
  int gd = (wd * 2 + id + sd) % 12;
  int gh = (wh * 8 + ih + sh) % 96;
  int gw = (ww * 8 + iw + sw) % 96;
  const float* src = x + (size_t)(gd * 9216 + gh * 96 + gw) * 120;
  float s = 0.f, s2 = 0.f;
  for (int c = 0; c < 120; ++c) { float v = src[c]; s += v; s2 += v * v; }
  float mean = s * (1.f / 120.f);
  float var  = s2 * (1.f / 120.f) - mean * mean;
  float rstd = rsqrtf(var + 1e-5f);
  size_t base = ((size_t)wi * 128 + tk) * 128;
  int p = tk & 63;
  float ye = (float)(((p >> 3) & 7) + 1) * (6.283185307f / 8.000001f);
  float xe = (float)((p & 7) + 1)        * (6.283185307f / 8.000001f);
  for (int c = 0; c < 128; ++c) {
    if (c < 120) {
      float v = (src[c] - mean) * rstd * w[c] + b[c];
      xw[base + c] = (half_t)v;
      int j = (c < 60) ? c : c - 60;
      float a2 = (c < 60) ? ye : xe;
      float tpw = powf(10000.f, (float)((j >> 1) * 2) / 60.f);
      float arg = a2 / tpw;
      float pv = (j & 1) ? cosf(arg) : sinf(arg);
      xwp[base + c] = (half_t)(v + pv);
    } else { xw[base + c] = (half_t)0.f; xwp[base + c] = (half_t)0.f; }
  }
}

__global__ void k_ln_tok(const float* __restrict__ x, const float* __restrict__ w,
                         const float* __restrict__ b, half_t* __restrict__ out) {
  int t = blockIdx.x * 128 + threadIdx.x;
  const float* src = x + (size_t)t * 120;
  float s = 0.f, s2 = 0.f;
  for (int c = 0; c < 120; ++c) { float v = src[c]; s += v; s2 += v * v; }
  float mean = s * (1.f / 120.f);
  float var  = s2 * (1.f / 120.f) - mean * mean;
  float rstd = rsqrtf(var + 1e-5f);
  size_t base = (size_t)t * 128;
  for (int c = 0; c < 128; ++c)
    out[base + c] = (c < 120) ? (half_t)((src[c] - mean) * rstd * w[c] + b[c]) : (half_t)0.f;
}

// ---------------- self attention (per-window, per-head) --------------------

__global__ __launch_bounds__(256) void k_self_attn(
    const half_t* __restrict__ xw, const half_t* __restrict__ wqkv,
    const float* __restrict__ rpb, half_t* __restrict__ cat, int use_mask) {
  extern __shared__ char smem[];
  float*  S_s  = (float*)smem;                  // 128*128 f32  (65536 B)
  half_t* xw_s = (half_t*)(smem + 65536);       // 128*128      (32768 B)
  half_t* qk_s = (half_t*)(smem + 98304);       // 128*64       (16384 B)
  half_t* vt_s = (half_t*)(smem + 114688);      // 32*128       (8192 B)
  half_t* P_s  = (half_t*)(smem + 122880);      // 128*128      (32768 B) -> 155648 total

  const int tid = threadIdx.x, lane = tid & 31, wv = tid >> 5;
  const int wi = blockIdx.x;
  const int wd = wi / 144, wh = (wi / 12) % 12, ww = wi % 12;
  const float scale = 0.2236067977f;  // 20^-0.5

  {
    const v8h* src = (const v8h*)(xw + (size_t)wi * 16384);
    v8h* dst = (v8h*)xw_s;
    for (int i = tid; i < 2048; i += 256) dst[i] = src[i];
  }
  __syncthreads();

  for (int h = 0; h < 6; ++h) {
    // --- QKV for this head: rows band wv, cols 0..95 (Q 0..31 K 32..63 V 64..95)
    {
      v8f acc[6];
      for (int cb = 0; cb < 6; ++cb) acc[cb] = vzero8();
      for (int kk = 0; kk < 128; kk += 32) {
        v16h a = frag_A(xw_s + (wv * 16) * 128 + kk, 128, lane);
        for (int cb = 0; cb < 6; ++cb) {
          v16h bfr = frag_B(wqkv + (size_t)(h * 96 + cb * 16) * 128 + kk, 128, lane);
          acc[cb] = wmma16(a, bfr, acc[cb]);
        }
      }
      int n = lane & 15, rb = (lane < 16) ? 0 : 8;
      for (int cb = 0; cb < 6; ++cb)
        for (int r = 0; r < 8; ++r) {
          int row = wv * 16 + rb + r;
          float v = acc[cb][r];
          if (cb < 2)      qk_s[row * 64 + cb * 16 + n] = (half_t)(v * scale);
          else if (cb < 4) qk_s[row * 64 + cb * 16 + n] = (half_t)v;
          else             vt_s[((cb - 4) * 16 + n) * 128 + row] = (half_t)v;
        }
    }
    __syncthreads();
    // --- fill S with rpb (+mask)
    for (int e = tid; e < 16384; e += 256) {
      int m = e >> 7, n = e & 127;
      int rd = (m >> 6) - (n >> 6) + 1;
      int rh = ((m >> 3) & 7) - ((n >> 3) & 7) + 7;
      int rw = (m & 7) - (n & 7) + 7;
      float v = rpb[((rd * 15 + rh) * 15 + rw) * 6 + h];
      if (use_mask) {
        int cm = group3(wd * 2 + (m >> 6), 10, 11) * 9 +
                 group3(wh * 8 + ((m >> 3) & 7), 88, 92) * 3 +
                 group3(ww * 8 + (m & 7), 88, 92);
        int cn = group3(wd * 2 + (n >> 6), 10, 11) * 9 +
                 group3(wh * 8 + ((n >> 3) & 7), 88, 92) * 3 +
                 group3(ww * 8 + (n & 7), 88, 92);
        if (cm != cn) v -= 100.f;
      }
      S_s[e] = v;
    }
    __syncthreads();
    // --- scores: S += (Q*scale) K^T  (hd=32 -> one k-step)
    {
      v16h a = frag_A(qk_s + (wv * 16) * 64, 64, lane);
      for (int cb = 0; cb < 8; ++cb) {
        v8f c = frag_C(S_s + (wv * 16) * 128 + cb * 16, 128, lane);
        v16h bfr = frag_B(qk_s + (cb * 16) * 64 + 32, 64, lane);
        c = wmma16(a, bfr, c);
        store_D(S_s + (wv * 16) * 128 + cb * 16, 128, lane, c);
      }
    }
    __syncthreads();
    // --- softmax over 128 cols
    if (tid < 128) {
      float* row = S_s + tid * 128;
      float mx = -1e30f;
      for (int j = 0; j < 128; ++j) mx = fmaxf(mx, row[j]);
      float sum = 0.f;
      for (int j = 0; j < 128; ++j) { float e2 = expf(row[j] - mx); row[j] = e2; sum += e2; }
      float inv = 1.f / sum;
      for (int j = 0; j < 128; ++j) P_s[tid * 128 + j] = (half_t)(row[j] * inv);
    }
    __syncthreads();
    // --- O = P V
    {
      v8f acc[2]; acc[0] = vzero8(); acc[1] = vzero8();
      for (int kk = 0; kk < 128; kk += 32) {
        v16h a = frag_A(P_s + (wv * 16) * 128 + kk, 128, lane);
        for (int cb = 0; cb < 2; ++cb) {
          v16h bfr = frag_B(vt_s + (cb * 16) * 128 + kk, 128, lane);
          acc[cb] = wmma16(a, bfr, acc[cb]);
        }
      }
      int n = lane & 15, rb = (lane < 16) ? 0 : 8;
      for (int cb = 0; cb < 2; ++cb) {
        int col = cb * 16 + n;
        if (col < 20)
          for (int r = 0; r < 8; ++r) {
            int t = wv * 16 + rb + r;
            cat[((size_t)wi * 128 + t) * 256 + 128 + h * 20 + col] = (half_t)acc[cb][r];
          }
      }
    }
    __syncthreads();
  }
  for (int e = tid; e < 1024; e += 256) {
    int t = e >> 3, c2 = 248 + (e & 7);
    cat[((size_t)wi * 128 + t) * 256 + c2] = (half_t)0.f;
  }
}

// ---------------- mutual attention (swap query halves, per-half keys) ------

__global__ __launch_bounds__(256) void k_mutual_attn(
    const half_t* __restrict__ xwp, const half_t* __restrict__ wqkv,
    half_t* __restrict__ cat, int use_mask) {
  extern __shared__ char smem[];
  float*  S_s  = (float*)smem;                  // 128*64 f32 (32768 B)
  half_t* xw_s = (half_t*)(smem + 32768);       // 32768 B
  half_t* qk_s = (half_t*)(smem + 65536);       // 16384 B
  half_t* vt_s = (half_t*)(smem + 81920);       // 8192 B
  half_t* P_s  = (half_t*)(smem + 90112);       // 16384 B -> 106496 total

  const int tid = threadIdx.x, lane = tid & 31, wv = tid >> 5;
  const int wi = blockIdx.x;
  const int wd = wi / 144, wh = (wi / 12) % 12, ww = wi % 12;
  const float scale = 0.2236067977f;

  {
    const v8h* src = (const v8h*)(xwp + (size_t)wi * 16384);
    v8h* dst = (v8h*)xw_s;
    for (int i = tid; i < 2048; i += 256) dst[i] = src[i];
  }
  __syncthreads();

  const int kb0 = (wv < 4) ? 0 : 64;  // key half for this band

  for (int h = 0; h < 6; ++h) {
    {
      v8f acc[6];
      for (int cb = 0; cb < 6; ++cb) acc[cb] = vzero8();
      for (int kk = 0; kk < 128; kk += 32) {
        v16h a = frag_A(xw_s + (wv * 16) * 128 + kk, 128, lane);
        for (int cb = 0; cb < 6; ++cb) {
          v16h bfr = frag_B(wqkv + (size_t)(h * 96 + cb * 16) * 128 + kk, 128, lane);
          acc[cb] = wmma16(a, bfr, acc[cb]);
        }
      }
      int n = lane & 15, rb = (lane < 16) ? 0 : 8;
      for (int cb = 0; cb < 6; ++cb)
        for (int r = 0; r < 8; ++r) {
          int row = wv * 16 + rb + r;
          float v = acc[cb][r];
          if (cb < 2)      qk_s[row * 64 + cb * 16 + n] = (half_t)(v * scale);
          else if (cb < 4) qk_s[row * 64 + cb * 16 + n] = (half_t)v;
          else             vt_s[((cb - 4) * 16 + n) * 128 + row] = (half_t)v;
        }
    }
    __syncthreads();
    // fill S (128 x 64) with first-frame mask (m2)
    for (int e = tid; e < 8192; e += 256) {
      int t = e >> 6, j = e & 63;
      float v = 0.f;
      if (use_mask) {
        int qn = t & 63;
        int cm = group3(wd * 2, 10, 11) * 9 +
                 group3(wh * 8 + ((qn >> 3) & 7), 88, 92) * 3 +
                 group3(ww * 8 + (qn & 7), 88, 92);
        int cn = group3(wd * 2, 10, 11) * 9 +
                 group3(wh * 8 + ((j >> 3) & 7), 88, 92) * 3 +
                 group3(ww * 8 + (j & 7), 88, 92);
        if (cm != cn) v = -100.f;
      }
      S_s[e] = v;
    }
    __syncthreads();
    // scores: query = Q[(t+64)%128] (swapped half), keys in own half
    {
      v16h a = frag_A(qk_s + (((wv + 4) & 7) * 16) * 64, 64, lane);
      for (int cb = 0; cb < 4; ++cb) {
        v8f c = frag_C(S_s + (wv * 16) * 64 + cb * 16, 64, lane);
        v16h bfr = frag_B(qk_s + (kb0 + cb * 16) * 64 + 32, 64, lane);
        c = wmma16(a, bfr, c);
        store_D(S_s + (wv * 16) * 64 + cb * 16, 64, lane, c);
      }
    }
    __syncthreads();
    if (tid < 128) {
      float* row = S_s + tid * 64;
      float mx = -1e30f;
      for (int j = 0; j < 64; ++j) mx = fmaxf(mx, row[j]);
      float sum = 0.f;
      for (int j = 0; j < 64; ++j) { float e2 = expf(row[j] - mx); row[j] = e2; sum += e2; }
      float inv = 1.f / sum;
      for (int j = 0; j < 64; ++j) P_s[tid * 64 + j] = (half_t)(row[j] * inv);
    }
    __syncthreads();
    {
      v8f acc[2]; acc[0] = vzero8(); acc[1] = vzero8();
      for (int kk = 0; kk < 64; kk += 32) {
        v16h a = frag_A(P_s + (wv * 16) * 64 + kk, 64, lane);
        for (int cb = 0; cb < 2; ++cb) {
          v16h bfr = frag_B(vt_s + (cb * 16) * 128 + kb0 + kk, 128, lane);
          acc[cb] = wmma16(a, bfr, acc[cb]);
        }
      }
      int n = lane & 15, rb = (lane < 16) ? 0 : 8;
      for (int cb = 0; cb < 2; ++cb) {
        int col = cb * 16 + n;
        if (col < 20)
          for (int r = 0; r < 8; ++r) {
            int t = wv * 16 + rb + r;
            cat[((size_t)wi * 128 + t) * 256 + h * 20 + col] = (half_t)acc[cb][r];
          }
      }
    }
    __syncthreads();
  }
  for (int e = tid; e < 1024; e += 256) {
    int t = e >> 3, c2 = 120 + (e & 7);
    cat[((size_t)wi * 128 + t) * 256 + c2] = (half_t)0.f;
  }
}

// ---------------- generic WMMA GEMM with fused epilogues --------------------

__global__ __launch_bounds__(256) void k_gemm(
    const half_t* __restrict__ A, int K, const half_t* __restrict__ W,
    const float* __restrict__ bias1, const float* __restrict__ bias2,
    float* __restrict__ outf, half_t* __restrict__ outh, int mode) {
  extern __shared__ char smem[];
  half_t* A_s = (half_t*)smem;
  half_t* W_s = A_s + 128 * K;
  const int tid = threadIdx.x, lane = tid & 31, wv = tid >> 5;
  const int m0 = blockIdx.x * 128;
  const int nb0 = blockIdx.y * 128;
  const int kv = K >> 3;
  for (int i = tid; i < 128 * kv; i += 256) {
    int row = i / kv, kc = (i % kv) * 8;
    *(v8h*)(A_s + row * K + kc) = *(const v8h*)(A + (size_t)(m0 + row) * K + kc);
    *(v8h*)(W_s + row * K + kc) = *(const v8h*)(W + (size_t)(nb0 + row) * K + kc);
  }
  __syncthreads();
  const int wm = wv >> 1, wn = wv & 1;
  v8f acc[2][4];
  for (int i = 0; i < 2; ++i) for (int j = 0; j < 4; ++j) acc[i][j] = vzero8();
  for (int kk = 0; kk < K; kk += 32) {
    v16h a0 = frag_A(A_s + (wm * 32) * K + kk, K, lane);
    v16h a1 = frag_A(A_s + (wm * 32 + 16) * K + kk, K, lane);
    for (int cb = 0; cb < 4; ++cb) {
      v16h bfr = frag_B(W_s + (wn * 64 + cb * 16) * K + kk, K, lane);
      acc[0][cb] = wmma16(a0, bfr, acc[0][cb]);
      acc[1][cb] = wmma16(a1, bfr, acc[1][cb]);
    }
  }
  const int n = lane & 15, rb = (lane < 16) ? 0 : 8;
  if (mode == GEMM_BIAS) {
    for (int rt = 0; rt < 2; ++rt)
      for (int cb = 0; cb < 4; ++cb) {
        int col = wn * 64 + cb * 16 + n;
        float bb = (col < 120) ? bias1[col] : 0.f;
        for (int r = 0; r < 8; ++r) {
          int m = m0 + wm * 32 + rt * 16 + rb + r;
          outf[(size_t)m * 128 + col] = acc[rt][cb][r] + bb;
        }
      }
  } else if (mode == GEMM_GATE) {
    for (int rt = 0; rt < 2; ++rt)
      for (int pr = 0; pr < 2; ++pr) {
        int hid = (blockIdx.y * 4 + wn * 2 + pr) * 16 + n;
        float b11 = (hid < 240) ? bias1[hid] : 0.f;
        float b12 = (hid < 240) ? bias2[hid] : 0.f;
        for (int r = 0; r < 8; ++r) {
          int m = m0 + wm * 32 + rt * 16 + rb + r;
          float x1 = acc[rt][2 * pr][r] + b11;
          float x2 = acc[rt][2 * pr + 1][r] + b12;
          float g = 0.5f * x1 * (1.f + erff(x1 * 0.7071067812f)) * x2;
          if (hid >= 240) g = 0.f;
          outh[(size_t)m * 256 + hid] = (half_t)g;
        }
      }
  } else {  // GEMM_RESBIAS: out += bias + acc into (M,120) residual buffer
    for (int rt = 0; rt < 2; ++rt)
      for (int cb = 0; cb < 4; ++cb) {
        int col = wn * 64 + cb * 16 + n;
        if (col < 120) {
          float bb = bias1[col];
          for (int r = 0; r < 8; ++r) {
            int m = m0 + wm * 32 + rt * 16 + rb + r;
            outf[(size_t)m * 120 + col] += acc[rt][cb][r] + bb;
          }
        }
      }
  }
}

// ---------------- window reverse + unshift + residual ----------------------

__global__ void k_unshift_add(float* __restrict__ x, const float* __restrict__ yw,
                              int sd, int sh, int sw) {
  int t = blockIdx.x * 128 + threadIdx.x;
  int d = t / 9216, hh = (t / 96) % 96, w2 = t % 96;
  int rd = (d - sd + 12) % 12, rh = (hh - sh + 96) % 96, rw = (w2 - sw + 96) % 96;
  int widx = (rd >> 1) * 144 + (rh >> 3) * 12 + (rw >> 3);
  int tk = (rd & 1) * 64 + (rh & 7) * 8 + (rw & 7);
  const float* src = yw + ((size_t)widx * 128 + tk) * 128;
  float* dst = x + (size_t)t * 120;
  for (int c = 0; c < 120; ++c) dst[c] += src[c];
}

// ---------------- host driver ----------------------------------------------

extern "C" void kernel_launch(void* const* d_in, const int* in_sizes, int n_in,
                              void* d_out, int out_size, void* d_ws, size_t ws_size,
                              hipStream_t stream) {
  (void)in_sizes; (void)n_in; (void)out_size; (void)ws_size;
  const float* x_in    = (const float*)d_in[0];
  const float* n1_w    = (const float*)d_in[1];
  const float* n1_b    = (const float*)d_in[2];
  const float* qkv_s_w = (const float*)d_in[3];
  const float* rpb     = (const float*)d_in[4];
  const float* qkv_m_w = (const float*)d_in[5];
  const float* proj_w  = (const float*)d_in[6];
  const float* proj_b  = (const float*)d_in[7];
  const float* n2_w    = (const float*)d_in[8];
  const float* n2_b    = (const float*)d_in[9];
  const float* fc11_w  = (const float*)d_in[10];
  const float* fc11_b  = (const float*)d_in[11];
  const float* fc12_w  = (const float*)d_in[12];
  const float* fc12_b  = (const float*)d_in[13];
  const float* fc2_w   = (const float*)d_in[14];
  const float* fc2_b   = (const float*)d_in[15];

  char* ws = (char*)d_ws;
  size_t o = 0;
  auto alloc = [&](size_t bytes) -> char* {
    char* p = ws + o; o += (bytes + 255) & ~(size_t)255; return p;
  };
  float*  x_tok    = (float*)alloc((size_t)NTOK * 120 * 4);     // 53 MB
  half_t* xw_h     = (half_t*)alloc((size_t)NTOK * 128 * 2);    // 28 MB (also h_h)
  half_t* xwp_h    = (half_t*)alloc((size_t)NTOK * 128 * 2);    // 28 MB
  half_t* attn_cat = (half_t*)alloc((size_t)NTOK * 256 * 2);    // 57 MB
  char*   big      = alloc((size_t)NTOK * 256 * 2);             // 57 MB shared
  float*  yw       = (float*)big;                               // NTOK x 128 f32
  half_t* hid_h    = (half_t*)big;                              // NTOK x 256 f16
  half_t* h_h      = xw_h;                                      // alias, live disjoint

  half_t *wqs[2], *wqm[2], *wpj[2], *w1c[2], *wf2[2];
  for (int l = 0; l < 2; ++l) {
    wqs[l] = (half_t*)alloc(576 * 128 * 2);
    wqm[l] = (half_t*)alloc(576 * 128 * 2);
    wpj[l] = (half_t*)alloc(128 * 256 * 2);
    w1c[l] = (half_t*)alloc(512 * 128 * 2);
    wf2[l] = (half_t*)alloc(128 * 256 * 2);
  }

  k_transpose_in<<<dim3(51840), dim3(256), 0, stream>>>(x_in, x_tok);
  for (int l = 0; l < 2; ++l) {
    k_pack_qkv<<<dim3(288), dim3(256), 0, stream>>>(qkv_s_w + (size_t)l * 360 * 120, wqs[l]);
    k_pack_qkv<<<dim3(288), dim3(256), 0, stream>>>(qkv_m_w + (size_t)l * 360 * 120, wqm[l]);
    k_pack_proj<<<dim3(128), dim3(256), 0, stream>>>(proj_w + (size_t)l * 120 * 240, wpj[l]);
    k_pack_w1c<<<dim3(256), dim3(256), 0, stream>>>(fc11_w + (size_t)l * 240 * 120,
                                                    fc12_w + (size_t)l * 240 * 120, w1c[l]);
    k_pack_fc2<<<dim3(128), dim3(256), 0, stream>>>(fc2_w + (size_t)l * 120 * 240, wf2[l]);
  }

  for (int l = 0; l < 2; ++l) {
    const int sd = l ? 1 : 0, sh = l ? 4 : 0, sw = l ? 4 : 0, um = l ? 1 : 0;
    k_ln_part<<<dim3(NWIN), dim3(128), 0, stream>>>(x_tok, n1_w + l * 120, n1_b + l * 120,
                                                    sd, sh, sw, xw_h, xwp_h);
    k_self_attn<<<dim3(NWIN), dim3(256), 155648, stream>>>(xw_h, wqs[l],
                                                           rpb + (size_t)l * 675 * 6,
                                                           attn_cat, um);
    k_mutual_attn<<<dim3(NWIN), dim3(256), 106496, stream>>>(xwp_h, wqm[l], attn_cat, um);
    k_gemm<<<dim3(NWIN, 1), dim3(256), 131072, stream>>>(attn_cat, 256, wpj[l],
                                                         proj_b + l * 120, nullptr,
                                                         yw, nullptr, GEMM_BIAS);
    k_unshift_add<<<dim3(NWIN), dim3(128), 0, stream>>>(x_tok, yw, sd, sh, sw);
    k_ln_tok<<<dim3(NWIN), dim3(128), 0, stream>>>(x_tok, n2_w + l * 120, n2_b + l * 120, h_h);
    k_gemm<<<dim3(NWIN, 4), dim3(256), 65536, stream>>>(h_h, 128, w1c[l],
                                                        fc11_b + l * 240, fc12_b + l * 240,
                                                        nullptr, hid_h, GEMM_GATE);
    k_gemm<<<dim3(NWIN, 1), dim3(256), 131072, stream>>>(hid_h, 256, wf2[l],
                                                         fc2_b + l * 120, nullptr,
                                                         x_tok, nullptr, GEMM_RESBIAS);
  }
  k_transpose_out<<<dim3(51840), dim3(256), 0, stream>>>(x_tok, (float*)d_out);
}
